// Head_66039417143860
// MI455X (gfx1250) — matrix operand order repaired and checked by reading, not compiled, not run on hardware
//
#include <hip/hip_runtime.h>
#include <hip/hip_bf16.h>

// ---------------------------------------------------------------------------
// Causal attention head with RoPE, B=256 T=256 C=H=500 (padded to 512).
// All GEMMs use v_wmma_f32_16x16x32_bf16 (bf16 inputs, f32 accumulate).
// Workspace layout (bf16 halves), ~270 MB total:
//   wt : 3*512*512      (W transposed+padded, K-contiguous rows)
//   xb : 65536*512      (x converted to bf16, padded)  -- ALIASED by pb later
//   qb : 65536*512      (q, RoPE'd in place)
//   kb : 65536*512      (k, RoPE'd in place)
//   vb : 256*512*256    (v stored TRANSPOSED: [b][h][t])
//   pb : 256*256*256    (softmax probabilities; aliases xb region)
// ---------------------------------------------------------------------------

typedef __attribute__((ext_vector_type(16))) __bf16 bf16x16;
typedef __attribute__((ext_vector_type(8)))  float  f32x8;

union FragU { uint4 q[2]; bf16x16 v; };

__device__ __forceinline__ unsigned short f2bf(float f) {
    unsigned int u = __float_as_uint(f);
    unsigned int r = u + 0x7fffu + ((u >> 16) & 1u);   // round-to-nearest-even
    return (unsigned short)(r >> 16);
}
__device__ __forceinline__ float bf2f(unsigned short h) {
    return __uint_as_float(((unsigned int)h) << 16);
}
__device__ __forceinline__ unsigned int pack2(float lo, float hi) {
    return (unsigned int)f2bf(lo) | ((unsigned int)f2bf(hi) << 16);
}

// A-matrix fragment (16x32 bf16): lane L%16 = row; K = {klo..klo+7, 16+klo..16+klo+7},
// klo = 8*(L>=16). p = row base + k-tile base (halves).
__device__ __forceinline__ bf16x16 load_a_frag(const unsigned short* p, int lane) {
    const int klo = (lane >= 16) ? 8 : 0;
    FragU f;
    f.q[0] = *reinterpret_cast<const uint4*>(p + klo);
    f.q[1] = *reinterpret_cast<const uint4*>(p + klo + 16);
    return f.v;
}
// B-matrix fragment (32x16 bf16): lane L%16 = column; K = {kb..kb+7, 16+kb..16+kb+7}, kb = 8*(L>=16).
__device__ __forceinline__ bf16x16 load_b_frag(const unsigned short* p, int lane) {
    const int kb = (lane >= 16) ? 8 : 0;
    FragU f;
    f.q[0] = *reinterpret_cast<const uint4*>(p + kb);
    f.q[1] = *reinterpret_cast<const uint4*>(p + kb + 16);
    return f.v;
}

__device__ __forceinline__ f32x8 wmma_bf16(bf16x16 a, bf16x16 b, f32x8 c) {
    return __builtin_amdgcn_wmma_f32_16x16x32_bf16(
        /*neg_a=*/false, a, /*neg_b=*/false, b,
        /*c_mod=*/(short)0, c, /*reuse_a=*/false, /*reuse_b=*/false);
}

__device__ __forceinline__ f32x8 zero8() {
    f32x8 z = {0.f, 0.f, 0.f, 0.f, 0.f, 0.f, 0.f, 0.f};
    return z;
}

// ---------------------------------------------------------------------------
// Kernel 0a: W[500][500] f32 -> wt[w][h][c] bf16 (transposed, padded 512x512)
// ---------------------------------------------------------------------------
__global__ __launch_bounds__(256) void k_prep_w(const float* __restrict__ Wq,
                                                const float* __restrict__ Wk,
                                                const float* __restrict__ Wv,
                                                unsigned short* __restrict__ wt) {
    int idx = blockIdx.x * 256 + threadIdx.x;       // < 3*512*512
    int w   = idx >> 18;
    int rem = idx & 262143;
    int h   = rem >> 9;
    int c   = rem & 511;
    const float* W = (w == 0) ? Wq : (w == 1) ? Wk : Wv;
    float v = (h < 500 && c < 500) ? W[c * 500 + h] : 0.f;
    wt[idx] = f2bf(v);
}

// ---------------------------------------------------------------------------
// Kernel 0b: x[65536][500] f32 -> xb[65536][512] bf16 (padded).
// One thread per 16 output columns.  grid = 65536*32/256 = 8192.
// ---------------------------------------------------------------------------
__global__ __launch_bounds__(256) void k_prep_x(const float* __restrict__ x,
                                                unsigned short* __restrict__ xb) {
    const int idx = blockIdx.x * 256 + threadIdx.x;   // < 65536*32
    const int row = idx >> 5;
    const int c0  = (idx & 31) * 16;
    const float*    src = x  + (size_t)row * 500;
    unsigned short* dst = xb + (size_t)row * 512 + c0;
    if (c0 + 16 <= 500) {
        float4 a = *reinterpret_cast<const float4*>(src + c0);
        float4 b = *reinterpret_cast<const float4*>(src + c0 + 4);
        float4 c = *reinterpret_cast<const float4*>(src + c0 + 8);
        float4 d = *reinterpret_cast<const float4*>(src + c0 + 12);
        uint4 o0, o1;
        o0.x = pack2(a.x, a.y); o0.y = pack2(a.z, a.w);
        o0.z = pack2(b.x, b.y); o0.w = pack2(b.z, b.w);
        o1.x = pack2(c.x, c.y); o1.y = pack2(c.z, c.w);
        o1.z = pack2(d.x, d.y); o1.w = pack2(d.z, d.w);
        *reinterpret_cast<uint4*>(dst)     = o0;
        *reinterpret_cast<uint4*>(dst + 8) = o1;
    } else {
#pragma unroll
        for (int j = 0; j < 16; ++j) {
            int c = c0 + j;
            dst[j] = f2bf((c < 500) ? src[c] : 0.f);
        }
    }
}

// ---------------------------------------------------------------------------
// Kernel 1: projections. q/k/v = x @ W.  Block: 128(M) x 128(N), K loop 512.
// grid = (512 mblocks, 4 nblocks, 3 matrices), 256 threads (8 waves).
// A and B fragments both loaded straight from global (K-contiguous bf16);
// no LDS, no barriers.  w==2 (v) is stored transposed into vb[b][h][t].
// ---------------------------------------------------------------------------
__global__ __launch_bounds__(256) void k_proj(const unsigned short* __restrict__ xb,
                                              const unsigned short* __restrict__ wt,
                                              unsigned short* __restrict__ qb,
                                              unsigned short* __restrict__ kb,
                                              unsigned short* __restrict__ vb) {
    const int tid   = threadIdx.x;
    const int lane  = tid & 31;
    const int wid   = tid >> 5;
    const int waveM = wid & 3;                // 0..3  -> 32-row strip
    const int waveN = wid >> 2;               // 0..1  -> 64-col strip
    const int mblock = blockIdx.x;            // 0..511
    const int nblock = blockIdx.y;            // 0..3
    const int w      = blockIdx.z;            // 0..2
    const unsigned short* wtw = wt + (size_t)w * 512 * 512;

    f32x8 acc[2][4];
#pragma unroll
    for (int mt = 0; mt < 2; ++mt)
#pragma unroll
        for (int nf = 0; nf < 4; ++nf) acc[mt][nf] = zero8();

    const unsigned short* arow[2];
#pragma unroll
    for (int mt = 0; mt < 2; ++mt) {
        const long r = (long)mblock * 128 + waveM * 32 + mt * 16 + (lane & 15);
        arow[mt] = xb + (size_t)r * 512;
    }
    const unsigned short* brow[4];
#pragma unroll
    for (int nf = 0; nf < 4; ++nf) {
        const int n = nblock * 128 + waveN * 64 + nf * 16 + (lane & 15);
        brow[nf] = wtw + (size_t)n * 512;
    }

    for (int ks = 0; ks < 16; ++ks) {
        const int kb0 = ks * 32;
        bf16x16 af[2], bfv[4];
#pragma unroll
        for (int mt = 0; mt < 2; ++mt) af[mt] = load_a_frag(arow[mt] + kb0, lane);
#pragma unroll
        for (int nf = 0; nf < 4; ++nf) bfv[nf] = load_b_frag(brow[nf] + kb0, lane);
#pragma unroll
        for (int mt = 0; mt < 2; ++mt)
#pragma unroll
            for (int nf = 0; nf < 4; ++nf)
                acc[mt][nf] = wmma_bf16(af[mt], bfv[nf], acc[mt][nf]);
    }

    const int lg = (lane >= 16) ? 8 : 0;
    if (w < 2) {
        unsigned short* out = (w == 0) ? qb : kb;
#pragma unroll
        for (int mt = 0; mt < 2; ++mt) {
            const long m0 = (long)mblock * 128 + waveM * 32 + mt * 16 + lg;
#pragma unroll
            for (int nf = 0; nf < 4; ++nf) {
                const int n = nblock * 128 + waveN * 64 + nf * 16 + (lane & 15);
#pragma unroll
                for (int r = 0; r < 8; ++r)
                    out[(size_t)(m0 + r) * 512 + n] = f2bf(acc[mt][nf][r]);
            }
        }
    } else {
        // transposed store: vb[(b*512 + h)*256 + t]; rows m0..m0+7 stay in one batch
#pragma unroll
        for (int mt = 0; mt < 2; ++mt) {
            const long m0 = (long)mblock * 128 + waveM * 32 + mt * 16 + lg;
            const int  b  = (int)(m0 >> 8);
            const int  t0 = (int)(m0 & 255);
#pragma unroll
            for (int nf = 0; nf < 4; ++nf) {
                const int h = nblock * 128 + waveN * 64 + nf * 16 + (lane & 15);
                unsigned short* dst = vb + ((size_t)b * 512 + h) * 256 + t0;
#pragma unroll
                for (int r = 0; r < 8; r += 2) {
                    *reinterpret_cast<unsigned int*>(dst + r) =
                        pack2(acc[mt][nf][r], acc[mt][nf][r + 1]);
                }
            }
        }
    }
}

// ---------------------------------------------------------------------------
// Kernel 2: in-place RoPE on qb, kb. One block per (b,t); thread i < 250.
// ---------------------------------------------------------------------------
__global__ __launch_bounds__(256) void k_rope(unsigned short* __restrict__ qb,
                                              unsigned short* __restrict__ kb) {
    const int bt = blockIdx.x;      // 0..65535
    const int i  = threadIdx.x;
    if (i >= 250) return;
    const int t = bt & 255;
    // freq = 1e-4 ^ ((i//2)/250) = exp(ln(1e-4) * (i//2)/250)
    const float freq = __expf(-9.210340371976184f * (float)(i >> 1) * (1.0f / 250.0f));
    const float ang  = (float)t * freq;
    float sn, cs;
    __sincosf(ang, &sn, &cs);
    const size_t base = (size_t)bt * 512;
    {
        float a = bf2f(qb[base + i]), b = bf2f(qb[base + i + 250]);
        qb[base + i]       = f2bf(a * cs - b * sn);
        qb[base + i + 250] = f2bf(a * sn + b * cs);
    }
    {
        float a = bf2f(kb[base + i]), b = bf2f(kb[base + i + 250]);
        kb[base + i]       = f2bf(a * cs - b * sn);
        kb[base + i + 250] = f2bf(a * sn + b * cs);
    }
}

// ---------------------------------------------------------------------------
// Kernel 3: scores + causal softmax.  Block: 64(M rows) x 256(cols), one batch.
// grid = (4 mtiles, 256 batches).  Waves: waveM in {0,1} x waveN in {0..3}.
// ---------------------------------------------------------------------------
__global__ __launch_bounds__(256) void k_scores(const unsigned short* __restrict__ qb,
                                                const unsigned short* __restrict__ kb,
                                                unsigned short* __restrict__ pb) {
    __shared__ float redmax[64][4];
    __shared__ float redsum[64][4];
    const int tid   = threadIdx.x;
    const int lane  = tid & 31;
    const int wid   = tid >> 5;
    const int waveM = wid >> 2;               // 0..1
    const int waveN = wid & 3;                // 0..3
    const int mtile = blockIdx.x;             // 0..3
    const int b     = blockIdx.y;             // 0..255

    const size_t qrow0 = ((size_t)b * 256 + mtile * 64) * 512;
    const size_t krow0 = (size_t)b * 256 * 512;

    f32x8 acc[2][4];
#pragma unroll
    for (int mt = 0; mt < 2; ++mt)
#pragma unroll
        for (int nf = 0; nf < 4; ++nf) acc[mt][nf] = zero8();

    for (int ks = 0; ks < 16; ++ks) {
        const int kb0 = ks * 32;
        bf16x16 af[2], bfv[4];
#pragma unroll
        for (int mt = 0; mt < 2; ++mt) {
            int r = waveM * 32 + mt * 16 + (lane & 15);
            af[mt] = load_a_frag(qb + qrow0 + (size_t)r * 512 + kb0, lane);
        }
#pragma unroll
        for (int nf = 0; nf < 4; ++nf) {
            int s = waveN * 64 + nf * 16 + (lane & 15);
            bfv[nf] = load_b_frag(kb + krow0 + (size_t)s * 512 + kb0, lane);
        }
#pragma unroll
        for (int mt = 0; mt < 2; ++mt)
#pragma unroll
            for (int nf = 0; nf < 4; ++nf)
                acc[mt][nf] = wmma_bf16(af[mt], bfv[nf], acc[mt][nf]);
    }

    const int   lg    = (lane >= 16) ? 8 : 0;
    const float scale = 0.04472135954999579f;   // 500^-0.5

    // scale + causal mask
#pragma unroll
    for (int mt = 0; mt < 2; ++mt)
#pragma unroll
        for (int nf = 0; nf < 4; ++nf)
#pragma unroll
            for (int r = 0; r < 8; ++r) {
                int t = mtile * 64 + waveM * 32 + mt * 16 + lg + r;
                int s = waveN * 64 + nf * 16 + (lane & 15);
                float v = acc[mt][nf][r] * scale;
                acc[mt][nf][r] = (s <= t) ? v : -__builtin_inff();
            }

    // partial row max (over this wave's 64 cols), then LDS across waveN
#pragma unroll
    for (int mt = 0; mt < 2; ++mt)
#pragma unroll
        for (int r = 0; r < 8; ++r) {
            float m = fmaxf(fmaxf(acc[mt][0][r], acc[mt][1][r]),
                            fmaxf(acc[mt][2][r], acc[mt][3][r]));
#pragma unroll
            for (int d = 8; d >= 1; d >>= 1) m = fmaxf(m, __shfl_xor(m, d, 32));
            if ((lane & 15) == 0) redmax[waveM * 32 + mt * 16 + lg + r][waveN] = m;
        }
    __syncthreads();

#pragma unroll
    for (int mt = 0; mt < 2; ++mt)
#pragma unroll
        for (int r = 0; r < 8; ++r) {
            const int rl = waveM * 32 + mt * 16 + lg + r;
            float fm = fmaxf(fmaxf(redmax[rl][0], redmax[rl][1]),
                             fmaxf(redmax[rl][2], redmax[rl][3]));
            float s = 0.f;
#pragma unroll
            for (int nf = 0; nf < 4; ++nf) {
                float e = __expf(acc[mt][nf][r] - fm);
                acc[mt][nf][r] = e;
                s += e;
            }
#pragma unroll
            for (int d = 8; d >= 1; d >>= 1) s += __shfl_xor(s, d, 32);
            if ((lane & 15) == 0) redsum[rl][waveN] = s;
        }
    __syncthreads();

#pragma unroll
    for (int mt = 0; mt < 2; ++mt)
#pragma unroll
        for (int r = 0; r < 8; ++r) {
            const int rl  = waveM * 32 + mt * 16 + lg + r;
            const float inv = 1.0f / (redsum[rl][0] + redsum[rl][1] +
                                      redsum[rl][2] + redsum[rl][3]);
            const int t = mtile * 64 + rl;
#pragma unroll
            for (int nf = 0; nf < 4; ++nf) {
                int s = waveN * 64 + nf * 16 + (lane & 15);
                pb[((size_t)b * 256 + t) * 256 + s] = f2bf(acc[mt][nf][r] * inv);
            }
        }
}

// ---------------------------------------------------------------------------
// Kernel 4: out = P @ V.  Block: 128(M=t) x 128(N=h).  K = 256 (=s).
// grid = (2 mblocks, 4 nblocks, 256 batches).
// ---------------------------------------------------------------------------
__global__ __launch_bounds__(256) void k_pv(const unsigned short* __restrict__ pb,
                                            const unsigned short* __restrict__ vb,
                                            float* __restrict__ out) {
    const int tid    = threadIdx.x;
    const int lane   = tid & 31;
    const int wid    = tid >> 5;
    const int waveM  = wid & 3;               // 0..3
    const int waveN  = wid >> 2;              // 0..1
    const int mblock = blockIdx.x;            // 0..1
    const int nblock = blockIdx.y;            // 0..3
    const int b      = blockIdx.z;            // 0..255

    f32x8 acc[2][4];
#pragma unroll
    for (int mt = 0; mt < 2; ++mt)
#pragma unroll
        for (int nf = 0; nf < 4; ++nf) acc[mt][nf] = zero8();

    for (int ks = 0; ks < 8; ++ks) {
        const int kb0 = ks * 32;
        bf16x16 af[2], bfv[4];
#pragma unroll
        for (int mt = 0; mt < 2; ++mt) {
            int t = mblock * 128 + waveM * 32 + mt * 16 + (lane & 15);
            af[mt] = load_a_frag(pb + ((size_t)b * 256 + t) * 256 + kb0, lane);
        }
#pragma unroll
        for (int nf = 0; nf < 4; ++nf) {
            int h = nblock * 128 + waveN * 64 + nf * 16 + (lane & 15);
            bfv[nf] = load_b_frag(vb + ((size_t)b * 512 + h) * 256 + kb0, lane);
        }
#pragma unroll
        for (int mt = 0; mt < 2; ++mt)
#pragma unroll
            for (int nf = 0; nf < 4; ++nf)
                acc[mt][nf] = wmma_bf16(af[mt], bfv[nf], acc[mt][nf]);
    }

    const int lg = (lane >= 16) ? 8 : 0;
#pragma unroll
    for (int mt = 0; mt < 2; ++mt) {
        const int t0 = mblock * 128 + waveM * 32 + mt * 16 + lg;
#pragma unroll
        for (int nf = 0; nf < 4; ++nf) {
            const int h = nblock * 128 + waveN * 64 + nf * 16 + (lane & 15);
            if (h < 500) {
#pragma unroll
                for (int r = 0; r < 8; ++r)
                    out[((size_t)b * 256 + t0 + r) * 500 + h] = acc[mt][nf][r];
            }
        }
    }
}

// ---------------------------------------------------------------------------
extern "C" void kernel_launch(void* const* d_in, const int* in_sizes, int n_in,
                              void* d_out, int out_size, void* d_ws, size_t ws_size,
                              hipStream_t stream) {
    const float* x  = (const float*)d_in[0];
    const float* Wq = (const float*)d_in[1];
    const float* Wk = (const float*)d_in[2];
    const float* Wv = (const float*)d_in[3];
    float* out = (float*)d_out;

    unsigned short* ws = (unsigned short*)d_ws;
    unsigned short* wt = ws;                                   //  3*512*512
    unsigned short* xb = wt + (size_t)3 * 512 * 512;           // 65536*512
    unsigned short* qb = xb + (size_t)65536 * 512;             // 65536*512
    unsigned short* kb = qb + (size_t)65536 * 512;             // 65536*512
    unsigned short* vb = kb + (size_t)65536 * 512;             // 256*512*256
    unsigned short* pb = xb;  // pb (256*256*256) aliases xb: xb dead after k_proj

    k_prep_w<<<3072, 256, 0, stream>>>(Wq, Wk, Wv, wt);
    k_prep_x<<<8192, 256, 0, stream>>>(x, xb);
    k_proj  <<<dim3(512, 4, 3), 256, 0, stream>>>(xb, wt, qb, kb, vb);
    k_rope  <<<65536, 256, 0, stream>>>(qb, kb);
    k_scores<<<dim3(4, 256), 256, 0, stream>>>(qb, kb, pb);
    k_pv    <<<dim3(2, 4, 256), 256, 0, stream>>>(pb, vb, out);
}
